// TverskyMultihead_72035191488660
// MI455X (gfx1250) — matrix operand order repaired and checked by reading, not compiled, not run on hardware
//
#include <hip/hip_runtime.h>
#include <hip/hip_bf16.h>
#include <stdint.h>

// ---- shapes (fixed by the reference) ----
#define NB 2048
#define NHEADS 16
#define DH 64
#define NF 4096
#define NP 512
#define TD 1024          // total_dim
#define OUTW 8192        // NHEADS * NP

#define LDK 72           // padded LDS row stride (ushorts), 144B rows (16B aligned)

typedef __attribute__((ext_vector_type(16))) __bf16 v16bf;
typedef __attribute__((ext_vector_type(16))) unsigned short v16us;
typedef __attribute__((ext_vector_type(8)))  float v8f;
typedef int v4i __attribute__((__vector_size__(16)));   // matches builtin param type

// ---- async global<->LDS (CDNA5): guarded so compile never breaks ----
#if __has_builtin(__builtin_amdgcn_global_load_async_to_lds_b128)
#define HAVE_ASYNC_LD 1
#else
#define HAVE_ASYNC_LD 0
#endif
#if __has_builtin(__builtin_amdgcn_global_store_async_from_lds_b128)
#define HAVE_ASYNC_ST 1
#else
#define HAVE_ASYNC_ST 0
#endif

#define G1P(p) ((__attribute__((address_space(1))) v4i*)(p))
#define L3P(p) ((__attribute__((address_space(3))) v4i*)(p))

__device__ __forceinline__ void async_wait() {
#if HAVE_ASYNC_LD || HAVE_ASYNC_ST
  asm volatile("s_wait_asynccnt 0x0" ::: "memory");
#endif
}

__device__ __forceinline__ unsigned short f32_to_bf16_raw(float f) {
  __bf16 b = (__bf16)f;                       // native cvt (RTNE) when available
  return __builtin_bit_cast(unsigned short, b);
}

__device__ __forceinline__ v8f wmma_bf16(v16bf a, v16bf b, v8f c) {
  return __builtin_amdgcn_wmma_f32_16x16x32_bf16(false, a, false, b, (short)0, c,
                                                 false, false);
}

// A fragment: 16x32 (MxK) bf16 from row-major LDS tile (ISA 7.12.2 layout).
__device__ __forceinline__ v16bf load_a_frag(const unsigned short* tile, int row0, int k0) {
  const int lane = threadIdx.x & 31;
  const unsigned short* p = tile + (row0 + (lane & 15)) * LDK + k0 + ((lane >> 4) << 3);
  v16us t;
#pragma unroll
  for (int i = 0; i < 8; ++i) t[i] = p[i];
#pragma unroll
  for (int i = 0; i < 8; ++i) t[8 + i] = p[16 + i];
  return __builtin_bit_cast(v16bf, t);
}

// B fragment: 32x16 (KxN) bf16 from row-major [N][K] LDS tile (B^T, contiguous K).
__device__ __forceinline__ v16bf load_b_frag(const unsigned short* tile, int n0, int k0) {
  const int lane = threadIdx.x & 31;
  const unsigned short* p = tile + (n0 + (lane & 15)) * LDK + k0 + ((lane >> 4) << 3);
  v16us t;
#pragma unroll
  for (int i = 0; i < 8; ++i) t[i] = p[i];
#pragma unroll
  for (int i = 0; i < 8; ++i) t[8 + i] = p[16 + i];
  return __builtin_bit_cast(v16bf, t);
}

// Stage a 64-row x 64-ushort tile from global (row stride in ushorts) into LDS.
__device__ __forceinline__ void stage_tile(const unsigned short* __restrict__ g,
                                           long long rowStride, unsigned short* lds) {
  const int t = threadIdx.x;
  const int row = t >> 2;
  const int seg = (t & 3) << 4;
  const unsigned short* gp = g + (long long)row * rowStride + seg;
  unsigned short* lp = lds + row * LDK + seg;
#if HAVE_ASYNC_LD
  __builtin_amdgcn_global_load_async_to_lds_b128(G1P(gp), L3P(lp), 0, 0);
  __builtin_amdgcn_global_load_async_to_lds_b128(G1P(gp + 8), L3P(lp + 8), 0, 0);
#else
  const uint4* src = reinterpret_cast<const uint4*>(gp);
  uint4* dst = reinterpret_cast<uint4*>(lp);
  dst[0] = src[0];
  dst[1] = src[1];
#endif
}

// ---------------- kernel A: f32 -> bf16 cast ----------------
__global__ void __launch_bounds__(256) cast_to_bf16(const float* __restrict__ src,
                                                    unsigned short* __restrict__ dst, int n) {
  int i = (blockIdx.x * 256 + threadIdx.x) * 4;
  if (i + 3 < n) {
    float4 v = *reinterpret_cast<const float4*>(src + i);
    ushort4 r;
    r.x = f32_to_bf16_raw(v.x);
    r.y = f32_to_bf16_raw(v.y);
    r.z = f32_to_bf16_raw(v.z);
    r.w = f32_to_bf16_raw(v.w);
    *reinterpret_cast<ushort4*>(dst + i) = r;
  }
}

// ---------------- kernel B: p-side precompute ----------------
// p_features[h,p,f] = sum_d proto[p,h*64+d] * feat[f,h*64+d]
// pw = relu(pf)^2, pp = relu(pf) (bf16), pwsum[h,p] = sum_f pw
__global__ void __launch_bounds__(256) tversky_pside(
    const unsigned short* __restrict__ protoB,   // [512][1024] bf16
    const unsigned short* __restrict__ featB,    // [4096][1024] bf16
    unsigned short* __restrict__ pw,             // [16][512][4096] bf16
    unsigned short* __restrict__ pp,             // [16][512][4096] bf16
    float* __restrict__ pwsum)                   // [16][512]
{
  __shared__ __align__(16) unsigned short sProto[64 * LDK];
  __shared__ __align__(16) unsigned short sFeat[64 * LDK];
  __shared__ __align__(16) unsigned short sW[64 * LDK];
  __shared__ __align__(16) unsigned short sP[64 * LDK];
  __shared__ float sRowSum[64];

  const int p0 = blockIdx.x * 64;
  const int h  = blockIdx.y;
  const int tid = threadIdx.x;
  const int lane = tid & 31;
  const int wv = tid >> 5;                 // wave32
  const int tm = wv >> 1;
  const int tn0 = (wv & 1) * 2;

  stage_tile(protoB + (long long)p0 * TD + h * DH, TD, sProto);
  async_wait();
  if (tid < 64) sRowSum[tid] = 0.0f;
  __syncthreads();

  float psum[8];
#pragma unroll
  for (int r = 0; r < 8; ++r) psum[r] = 0.0f;

  for (int fc = 0; fc < NF; fc += 64) {
    stage_tile(featB + (long long)fc * TD + h * DH, TD, sFeat);
    async_wait();
    __syncthreads();

    v16bf a0 = load_a_frag(sProto, tm * 16, 0);
    v16bf a1 = load_a_frag(sProto, tm * 16, 32);
#pragma unroll
    for (int t = 0; t < 2; ++t) {
      const int tn = tn0 + t;
      v16bf b0 = load_b_frag(sFeat, tn * 16, 0);
      v16bf b1 = load_b_frag(sFeat, tn * 16, 32);
      v8f acc = {};
      acc = wmma_bf16(a0, b0, acc);
      acc = wmma_bf16(a1, b1, acc);
      const int mbase = tm * 16 + ((lane >> 4) << 3);
      const int n = tn * 16 + (lane & 15);
#pragma unroll
      for (int r = 0; r < 8; ++r) {
        float v = acc[r];
        float pres = v > 0.0f ? v : 0.0f;
        float w = pres * pres;
        sP[(mbase + r) * LDK + n] = f32_to_bf16_raw(pres);
        sW[(mbase + r) * LDK + n] = f32_to_bf16_raw(w);
        psum[r] += w;
      }
    }
    __syncthreads();
    {
      const int row = tid >> 2;
      const int seg = (tid & 3) << 4;
      const long long gofs = ((long long)(h * NP + p0 + row)) * NF + fc + seg;
      unsigned short* gw = pw + gofs;
      unsigned short* gp = pp + gofs;
      const unsigned short* sw = sW + row * LDK + seg;
      const unsigned short* sp = sP + row * LDK + seg;
#if HAVE_ASYNC_ST
      __builtin_amdgcn_global_store_async_from_lds_b128(G1P(gw), L3P(sw), 0, 0);
      __builtin_amdgcn_global_store_async_from_lds_b128(G1P(gw + 8), L3P(sw + 8), 0, 0);
      __builtin_amdgcn_global_store_async_from_lds_b128(G1P(gp), L3P(sp), 0, 0);
      __builtin_amdgcn_global_store_async_from_lds_b128(G1P(gp + 8), L3P(sp + 8), 0, 0);
#else
      reinterpret_cast<uint4*>(gw)[0] = reinterpret_cast<const uint4*>(sw)[0];
      reinterpret_cast<uint4*>(gw)[1] = reinterpret_cast<const uint4*>(sw)[1];
      reinterpret_cast<uint4*>(gp)[0] = reinterpret_cast<const uint4*>(sp)[0];
      reinterpret_cast<uint4*>(gp)[1] = reinterpret_cast<const uint4*>(sp)[1];
#endif
    }
    async_wait();   // sW/sP must be drained before next-chunk overwrite
    __syncthreads();
  }

  {
    const int mbase = tm * 16 + ((lane >> 4) << 3);
#pragma unroll
    for (int r = 0; r < 8; ++r) atomicAdd(&sRowSum[mbase + r], psum[r]);
  }
  __syncthreads();
  if (tid < 64) pwsum[h * NP + p0 + tid] = sRowSum[tid];
}

// ---------------- kernel C: fused x-side + 3 output GEMMs ----------------
__global__ void __launch_bounds__(256) tversky_main(
    const unsigned short* __restrict__ xB,       // [2048][1024] bf16
    const unsigned short* __restrict__ featB,    // [4096][1024] bf16
    const unsigned short* __restrict__ pw,       // [16][512][4096] bf16
    const unsigned short* __restrict__ pp,       // [16][512][4096] bf16
    const float* __restrict__ pwsum,             // [16][512]
    const float* __restrict__ theta,
    const float* __restrict__ alpha,
    const float* __restrict__ beta,
    float* __restrict__ out)                     // [2048][8192]
{
  __shared__ __align__(16) unsigned short sX[64 * LDK];
  __shared__ __align__(16) unsigned short sFeatD[2][64 * LDK];
  __shared__ __align__(16) unsigned short sPWD[2][64 * LDK];
  __shared__ __align__(16) unsigned short sPPD[2][64 * LDK];
  __shared__ __align__(16) unsigned short sXW[64 * LDK];
  __shared__ __align__(16) unsigned short sXP[64 * LDK];
  __shared__ float sRowSum[64];

  const int b0 = blockIdx.x * 64;
  const int p0 = blockIdx.y * 64;
  const int h  = blockIdx.z;
  const int tid = threadIdx.x;
  const int lane = tid & 31;
  const int wv = tid >> 5;                 // 8 waves of 32
  const int tm = wv >> 1;                  // stage-1: 4x2 waves over 4x4 tiles
  const int tn0 = (wv & 1) * 2;
  const int wm = wv >> 1;                  // stage-2: rows wm*16.., cols (wv&1)*32..
  const int wn = (wv & 1) * 32;

  const long long pbase = ((long long)(h * NP + p0)) * NF;
  const unsigned short* featBase = featB + h * DH;

  // prologue: stage x-tile and chunk 0 into buffer 0
  stage_tile(xB + (long long)b0 * TD + h * DH, TD, sX);
  stage_tile(featBase, TD, sFeatD[0]);
  stage_tile(pw + pbase, NF, sPWD[0]);
  stage_tile(pp + pbase, NF, sPPD[0]);
  async_wait();
  if (tid < 64) sRowSum[tid] = 0.0f;
  __syncthreads();

  v8f c0 = {}, c1 = {}, xi0 = {}, xi1 = {}, pi0 = {}, pi1 = {};
  float psum[8];
#pragma unroll
  for (int r = 0; r < 8; ++r) psum[r] = 0.0f;

  for (int it = 0; it < NF / 64; ++it) {
    const int fc = it * 64;
    const int cur = it & 1;
    const int nxt = cur ^ 1;

    // overlap: issue next chunk's staging while computing this one
    if (it + 1 < NF / 64) {
      stage_tile(featBase + (long long)(fc + 64) * TD, TD, sFeatD[nxt]);
      stage_tile(pw + pbase + fc + 64, NF, sPWD[nxt]);
      stage_tile(pp + pbase + fc + 64, NF, sPPD[nxt]);
    }

    const unsigned short* sFeat = sFeatD[cur];
    const unsigned short* sPW = sPWD[cur];
    const unsigned short* sPP = sPPD[cur];

    // ---- stage 1: x_features chunk -> relu / relu^2 into LDS (bf16) ----
    {
      v16bf a0 = load_a_frag(sX, tm * 16, 0);
      v16bf a1 = load_a_frag(sX, tm * 16, 32);
#pragma unroll
      for (int t = 0; t < 2; ++t) {
        const int tn = tn0 + t;
        v16bf b0 = load_b_frag(sFeat, tn * 16, 0);
        v16bf b1 = load_b_frag(sFeat, tn * 16, 32);
        v8f acc = {};
        acc = wmma_bf16(a0, b0, acc);
        acc = wmma_bf16(a1, b1, acc);
        const int mbase = tm * 16 + ((lane >> 4) << 3);
        const int n = tn * 16 + (lane & 15);
#pragma unroll
        for (int r = 0; r < 8; ++r) {
          float v = acc[r];
          float pres = v > 0.0f ? v : 0.0f;
          float w = pres * pres;
          sXP[(mbase + r) * LDK + n] = f32_to_bf16_raw(pres);
          sXW[(mbase + r) * LDK + n] = f32_to_bf16_raw(w);
          psum[r] += w;
        }
      }
    }
    __syncthreads();

    // ---- stage 2: accumulate the three output GEMMs over this f-chunk ----
    {
      v16bf aw0 = load_a_frag(sXW, wm * 16, 0);
      v16bf aw1 = load_a_frag(sXW, wm * 16, 32);
      v16bf ap0 = load_a_frag(sXP, wm * 16, 0);
      v16bf ap1 = load_a_frag(sXP, wm * 16, 32);
      {
        v16bf bw0 = load_b_frag(sPW, wn, 0);
        v16bf bw1 = load_b_frag(sPW, wn, 32);
        v16bf bp0 = load_b_frag(sPP, wn, 0);
        v16bf bp1 = load_b_frag(sPP, wn, 32);
        c0  = wmma_bf16(aw0, bw0, c0);  c0  = wmma_bf16(aw1, bw1, c0);
        xi0 = wmma_bf16(aw0, bp0, xi0); xi0 = wmma_bf16(aw1, bp1, xi0);
        pi0 = wmma_bf16(ap0, bw0, pi0); pi0 = wmma_bf16(ap1, bw1, pi0);
      }
      {
        v16bf bw0 = load_b_frag(sPW, wn + 16, 0);
        v16bf bw1 = load_b_frag(sPW, wn + 16, 32);
        v16bf bp0 = load_b_frag(sPP, wn + 16, 0);
        v16bf bp1 = load_b_frag(sPP, wn + 16, 32);
        c1  = wmma_bf16(aw0, bw0, c1);  c1  = wmma_bf16(aw1, bw1, c1);
        xi1 = wmma_bf16(aw0, bp0, xi1); xi1 = wmma_bf16(aw1, bp1, xi1);
        pi1 = wmma_bf16(ap0, bw0, pi1); pi1 = wmma_bf16(ap1, bw1, pi1);
      }
    }
    async_wait();      // next chunk's async copies landed (overlapped with WMMA)
    __syncthreads();   // also fences sXW/sXP reuse and buffer swap
  }

  // x_weighted_sum reduction (rows local to this block: full sum over F)
  {
    const int mbase = tm * 16 + ((lane >> 4) << 3);
#pragma unroll
    for (int r = 0; r < 8; ++r) atomicAdd(&sRowSum[mbase + r], psum[r]);
  }
  __syncthreads();

  const float th = theta[h];
  const float al = alpha[h];
  const float be = beta[h];
  const int mbase = wm * 16 + ((lane >> 4) << 3);
#pragma unroll
  for (int t = 0; t < 2; ++t) {
    const int n = wn + t * 16 + (lane & 15);
    const float ps = pwsum[h * NP + p0 + n];
    const v8f cc = t ? c1 : c0;
    const v8f xx = t ? xi1 : xi0;
    const v8f pp2 = t ? pi1 : pi0;
#pragma unroll
    for (int r = 0; r < 8; ++r) {
      const int m = mbase + r;
      const float xs = sRowSum[m];
      float v = th * cc[r] - al * (xs - xx[r]) - be * (ps - pp2[r]);
      out[(long long)(b0 + m) * OUTW + h * NP + p0 + n] = v;
    }
  }
}

// ---------------- host launcher ----------------
extern "C" void kernel_launch(void* const* d_in, const int* in_sizes, int n_in,
                              void* d_out, int out_size, void* d_ws, size_t ws_size,
                              hipStream_t stream) {
  const float* x          = (const float*)d_in[0];
  const float* features   = (const float*)d_in[1];
  const float* prototypes = (const float*)d_in[2];
  const float* theta      = (const float*)d_in[3];
  const float* alpha      = (const float*)d_in[4];
  const float* beta       = (const float*)d_in[5];
  float* out = (float*)d_out;

  char* ws = (char*)d_ws;
  const size_t xBytes  = (size_t)NB * TD * 2;           //   4 MB
  const size_t fBytes  = (size_t)NF * TD * 2;           //   8 MB
  const size_t pBytes  = (size_t)NP * TD * 2;           //   1 MB
  const size_t pwBytes = (size_t)NHEADS * NP * NF * 2;  //  64 MB
  unsigned short* xB     = (unsigned short*)(ws);
  unsigned short* featB  = (unsigned short*)(ws + xBytes);
  unsigned short* protoB = (unsigned short*)(ws + xBytes + fBytes);
  unsigned short* pwBuf  = (unsigned short*)(ws + xBytes + fBytes + pBytes);
  unsigned short* ppBuf  = (unsigned short*)(ws + xBytes + fBytes + pBytes + pwBytes);
  float* pwsum           = (float*)(ws + xBytes + fBytes + pBytes + 2 * pwBytes);

  const int nx = NB * TD, nf = NF * TD, np = NP * TD;
  cast_to_bf16<<<(nx / 4 + 255) / 256, 256, 0, stream>>>(x, xB, nx);
  cast_to_bf16<<<(nf / 4 + 255) / 256, 256, 0, stream>>>(features, featB, nf);
  cast_to_bf16<<<(np / 4 + 255) / 256, 256, 0, stream>>>(prototypes, protoB, np);

  tversky_pside<<<dim3(NP / 64, NHEADS), 256, 0, stream>>>(protoB, featB, pwBuf, ppBuf, pwsum);

  tversky_main<<<dim3(NB / 64, NP / 64, NHEADS), 256, 0, stream>>>(
      xB, featB, pwBuf, ppBuf, pwsum, theta, alpha, beta, out);
}
